// QwenStyleSparseMoEBlock_83597243449391
// MI455X (gfx1250) — matrix operand order repaired and checked by reading, not compiled, not run on hardware
//
#include <hip/hip_runtime.h>
#include <hip/hip_bf16.h>

// MoE block (Qwen-style sparse MoE), MI455X / gfx1250, wave32 + WMMA bf16.
//
// Pipeline per launch:
//   1) router -> cw[T,4]
//   2) x fp32 -> bf16
//   3) weights fp32 [K][N] -> bf16 transposed [N][K] (BW-trivial pre-pass)
//   4) gate/up GEMM (wmma bf16) + erf-GELU*up -> h bf16
//   5) down GEMM + per-token top-2 weighted combine -> out fp32
// GEMM tiles are double-buffered in LDS and filled with gfx1250 async
// global->LDS b128 copies (ASYNCcnt), overlapping copy with WMMA compute.
//
// Workspace: cw 64KB | xbf 8MB | h 64MB | wg_t 16MB | wu_t 16MB | wd_t 16MB.

#define T_TOK 4096
#define DM    1024
#define DF    2048
#define NE    4

typedef __attribute__((ext_vector_type(16))) __bf16 bf16x16;
typedef __attribute__((ext_vector_type(8)))  float  f32x8;

union Frag { uint4 q[2]; bf16x16 v; };

__device__ __forceinline__ unsigned short f2bf(float f) {
  unsigned int u = __float_as_uint(f);
  unsigned int r = u + 0x7FFFu + ((u >> 16) & 1u);   // round-to-nearest-even
  if ((u & 0x7F800000u) == 0x7F800000u) r = u;       // inf/NaN passthrough
  return (unsigned short)(r >> 16);
}

__device__ __forceinline__ float gelu_erf(float x) {
  return 0.5f * x * (1.0f + erff(x * 0.7071067811865475f));
}

// LDS byte address = low 32 bits of the generic pointer (aperture rules).
__device__ __forceinline__ unsigned lds_addr(const void* l) {
  return (unsigned)(unsigned long long)l;
}

// gfx1250 async global->LDS 128-bit copy, tracked by ASYNCcnt.
__device__ __forceinline__ void async_b128(unsigned lds, const void* g) {
  asm volatile("global_load_async_to_lds_b128 %0, %1, off"
               :: "v"(lds), "v"(g) : "memory");
}

__device__ __forceinline__ void wait_async0() {
#if __has_builtin(__builtin_amdgcn_s_wait_asynccnt)
  __builtin_amdgcn_s_wait_asynccnt(0);
#else
  asm volatile("s_wait_asynccnt 0x0" ::: "memory");
#endif
}

// ---------------------------------------------------------------------------
// 1) Router: logits = x @ w_gate^T ; softmax ; top-2 ; renorm -> cw[T, 4]
// ---------------------------------------------------------------------------
__global__ __launch_bounds__(256) void moe_router(const float* __restrict__ x,
                                                  const float* __restrict__ wg,
                                                  float* __restrict__ cw) {
  const int wave = threadIdx.x >> 5;
  const int lane = threadIdx.x & 31;
  const int tok  = blockIdx.x * 8 + wave;
  if (tok >= T_TOK) return;
  const float* xr = x + (size_t)tok * DM;
  float s0 = 0.f, s1 = 0.f, s2 = 0.f, s3 = 0.f;
  for (int k = lane; k < DM; k += 32) {
    const float xv = xr[k];
    s0 += xv * wg[0 * DM + k];
    s1 += xv * wg[1 * DM + k];
    s2 += xv * wg[2 * DM + k];
    s3 += xv * wg[3 * DM + k];
  }
  for (int off = 16; off > 0; off >>= 1) {
    s0 += __shfl_xor(s0, off, 32);
    s1 += __shfl_xor(s1, off, 32);
    s2 += __shfl_xor(s2, off, 32);
    s3 += __shfl_xor(s3, off, 32);
  }
  if (lane == 0) {
    float l[4] = {s0, s1, s2, s3};
    float mx = fmaxf(fmaxf(l[0], l[1]), fmaxf(l[2], l[3]));
    float p[4]; float den = 0.f;
    for (int e = 0; e < 4; ++e) { p[e] = expf(l[e] - mx); den += p[e]; }
    for (int e = 0; e < 4; ++e) p[e] /= den;
    int i1 = 0;
    for (int e = 1; e < 4; ++e) if (p[e] > p[i1]) i1 = e;   // ties -> lowest idx
    int i2 = -1;
    for (int e = 0; e < 4; ++e) if (e != i1 && (i2 < 0 || p[e] > p[i2])) i2 = e;
    const float tot = p[i1] + p[i2];
    for (int e = 0; e < 4; ++e) {
      float w = (e == i1) ? p[i1] / tot : (e == i2) ? p[i2] / tot : 0.f;
      cw[tok * 4 + e] = w;
    }
  }
}

// ---------------------------------------------------------------------------
// 2) x fp32 -> bf16 packed pairs.
// ---------------------------------------------------------------------------
__global__ __launch_bounds__(256) void cvt_bf16(const float* __restrict__ x,
                                                unsigned int* __restrict__ o,
                                                int n2) {
  const int i = blockIdx.x * blockDim.x + threadIdx.x;
  if (i >= n2) return;
  const unsigned int lo = f2bf(x[2 * i]);
  const unsigned int hi = f2bf(x[2 * i + 1]);
  o[i] = lo | (hi << 16);
}

// ---------------------------------------------------------------------------
// 3) Weight convert + transpose: src fp32 [E][K][N] -> dst bf16 [E][N][K].
// ---------------------------------------------------------------------------
__global__ __launch_bounds__(256) void wt_cvt_t(const float* __restrict__ src,
                                                unsigned short* __restrict__ dst,
                                                int K, int N) {
  __shared__ unsigned short tile[64 * 65];
  const int n0 = blockIdx.x * 64;
  const int k0 = blockIdx.y * 64;
  const int e  = blockIdx.z;
  const float* srcE = src + (size_t)e * K * N;
  unsigned short* dstE = dst + (size_t)e * K * N;
  const int tid = threadIdx.x;
  for (int j = 0; j < 16; ++j) {
    const int i  = tid + 256 * j;
    const int kk = i >> 6;
    const int nn = i & 63;
    tile[nn * 65 + kk] = f2bf(srcE[(size_t)(k0 + kk) * N + (n0 + nn)]);
  }
  __syncthreads();
  for (int j = 0; j < 16; ++j) {
    const int i  = tid + 256 * j;
    const int nn = i >> 6;
    const int kk = i & 63;
    dstE[(size_t)(n0 + nn) * K + (k0 + kk)] = tile[nn * 65 + kk];
  }
}

// ---------------------------------------------------------------------------
// 4) Gate/Up GEMM + exact-erf GELU-GLU.  Block tile 64(tok) x 128(F),
//    8 wave32s (2x4), wave tile 32x32, K step 32.  Double-buffered LDS,
//    async fills overlap WMMA compute; one barrier per K step.
// ---------------------------------------------------------------------------
__global__ __launch_bounds__(256) void moe_gateup(
    const unsigned short* __restrict__ xbf,
    const unsigned short* __restrict__ wgt,   // bf16 [E][F][D]  (N-major)
    const unsigned short* __restrict__ wut,   // bf16 [E][F][D]
    unsigned short* __restrict__ h) {
  __shared__ __align__(16) unsigned short As[2][64 * 40];
  __shared__ __align__(16) unsigned short Bg[2][128 * 40];
  __shared__ __align__(16) unsigned short Bu[2][128 * 40];

  const int n0 = blockIdx.x * 128;
  const int t0 = blockIdx.y * 64;
  const int e  = blockIdx.z;
  const size_t eofft = (size_t)e * DM * DF;

  const int tid  = threadIdx.x;
  const int lane = tid & 31;
  const int wave = tid >> 5;
  const int wm   = wave >> 2;
  const int wn   = wave & 3;
  const int half = lane >> 4;
  const int lr   = lane & 15;

  // Per-thread async copy endpoints (advance by 32 bf16 per K step).
  const int ma = tid >> 2, sega = tid & 3;       // A: 64 rows x 4 chunks
  const int nb = tid >> 2, segb = tid & 3;       // B: rows nb, nb+64
  const unsigned short* aS  = xbf + (size_t)(t0 + ma) * DM + sega * 8;
  const unsigned short* gS0 = wgt + eofft + (size_t)(n0 + nb) * DM + segb * 8;
  const unsigned short* gS1 = gS0 + (size_t)64 * DM;
  const unsigned short* uS0 = wut + eofft + (size_t)(n0 + nb) * DM + segb * 8;
  const unsigned short* uS1 = uS0 + (size_t)64 * DM;
  unsigned aD[2], gD0[2], gD1[2], uD0[2], uD1[2];
  for (int b = 0; b < 2; ++b) {
    aD[b]  = lds_addr(&As[b][ma * 40 + sega * 8]);
    gD0[b] = lds_addr(&Bg[b][nb * 40 + segb * 8]);
    gD1[b] = lds_addr(&Bg[b][(nb + 64) * 40 + segb * 8]);
    uD0[b] = lds_addr(&Bu[b][nb * 40 + segb * 8]);
    uD1[b] = lds_addr(&Bu[b][(nb + 64) * 40 + segb * 8]);
  }

  f32x8 gacc[2][2], uacc[2][2];
  for (int i = 0; i < 2; ++i)
    for (int j = 0; j < 2; ++j)
      for (int r = 0; r < 8; ++r) { gacc[i][j][r] = 0.f; uacc[i][j][r] = 0.f; }

  const int NIT = DM / 32;
  // Prologue fill of buffer 0.
  {
    async_b128(aD[0], aS);
    async_b128(gD0[0], gS0);
    async_b128(gD1[0], gS1);
    async_b128(uD0[0], uS0);
    async_b128(uD1[0], uS1);
  }

  for (int it = 0; it < NIT; ++it) {
    wait_async0();         // this wave's fill of buf[it&1] complete
    __syncthreads();       // all waves' fills visible; old buffer free
    if (it + 1 < NIT) {
      const int koff = (it + 1) * 32;
      const int b = (it + 1) & 1;
      async_b128(aD[b],  aS  + koff);
      async_b128(gD0[b], gS0 + koff);
      async_b128(gD1[b], gS1 + koff);
      async_b128(uD0[b], uS0 + koff);
      async_b128(uD1[b], uS1 + koff);
    }
    const int b = it & 1;
    Frag a[2], bg[2], bu[2];
    for (int sm = 0; sm < 2; ++sm) {             // A layout (16x32 bf16)
      const int m = wm * 32 + sm * 16 + lr;
      a[sm].q[0] = *(const uint4*)&As[b][m * 40 + half * 8];
      a[sm].q[1] = *(const uint4*)&As[b][m * 40 + 16 + half * 8];
    }
    for (int sn = 0; sn < 2; ++sn) {             // B layout (32x16 bf16)
      const int n = wn * 32 + sn * 16 + lr;
      bg[sn].q[0] = *(const uint4*)&Bg[b][n * 40 + half * 8];
      bg[sn].q[1] = *(const uint4*)&Bg[b][n * 40 + 16 + half * 8];
      bu[sn].q[0] = *(const uint4*)&Bu[b][n * 40 + half * 8];
      bu[sn].q[1] = *(const uint4*)&Bu[b][n * 40 + 16 + half * 8];
    }
    for (int sm = 0; sm < 2; ++sm)
      for (int sn = 0; sn < 2; ++sn) {
        gacc[sm][sn] = __builtin_amdgcn_wmma_f32_16x16x32_bf16(
            false, a[sm].v, false, bg[sn].v, (short)0, gacc[sm][sn], false, false);
        uacc[sm][sn] = __builtin_amdgcn_wmma_f32_16x16x32_bf16(
            false, a[sm].v, false, bu[sn].v, (short)0, uacc[sm][sn], false, false);
      }
  }

  for (int sm = 0; sm < 2; ++sm)
    for (int sn = 0; sn < 2; ++sn)
      for (int r = 0; r < 8; ++r) {
        const int row = t0 + wm * 32 + sm * 16 + r + half * 8;
        const int col = n0 + wn * 32 + sn * 16 + lr;
        const float g = gacc[sm][sn][r];
        const float u = uacc[sm][sn][r];
        h[((size_t)e * T_TOK + row) * DF + col] = f2bf(gelu_erf(g) * u);
      }
}

// ---------------------------------------------------------------------------
// 5) Down-proj + weighted top-2 combine.  Block tile 64(tok) x 128(D).
//    Flattened (expert, k-tile) pipeline; tacc folded into facc with cw[t,e]
//    at each expert boundary (uniform branch).
// ---------------------------------------------------------------------------
__global__ __launch_bounds__(256) void moe_down(
    const unsigned short* __restrict__ h,     // bf16 [E][T][F]
    const unsigned short* __restrict__ wdt,   // bf16 [E][D][F]  (N-major)
    const float* __restrict__ cw,
    float* __restrict__ out) {
  __shared__ __align__(16) unsigned short As[2][64 * 40];
  __shared__ __align__(16) unsigned short Bs[2][128 * 40];
  __shared__ float cwl[64 * 4];

  const int n0 = blockIdx.x * 128;   // D columns
  const int t0 = blockIdx.y * 64;    // tokens

  const int tid  = threadIdx.x;
  const int lane = tid & 31;
  const int wave = tid >> 5;
  const int wm   = wave >> 2;
  const int wn   = wave & 3;
  const int half = lane >> 4;
  const int lr   = lane & 15;

  cwl[tid] = cw[(t0 + (tid >> 2)) * 4 + (tid & 3)];

  const int ma = tid >> 2, sega = tid & 3;
  const int nb = tid >> 2, segb = tid & 3;
  const unsigned short* aS  = h + (size_t)(t0 + ma) * DF + sega * 8;
  const unsigned short* bS0 = wdt + (size_t)(n0 + nb) * DF + segb * 8;
  const unsigned short* bS1 = bS0 + (size_t)64 * DF;
  unsigned aD[2], bD0[2], bD1[2];
  for (int b = 0; b < 2; ++b) {
    aD[b]  = lds_addr(&As[b][ma * 40 + sega * 8]);
    bD0[b] = lds_addr(&Bs[b][nb * 40 + segb * 8]);
    bD1[b] = lds_addr(&Bs[b][(nb + 64) * 40 + segb * 8]);
  }

  f32x8 facc[2][2], tacc[2][2];
  for (int i = 0; i < 2; ++i)
    for (int j = 0; j < 2; ++j)
      for (int r = 0; r < 8; ++r) { facc[i][j][r] = 0.f; tacc[i][j][r] = 0.f; }

  const int KT   = DF / 32;          // 64 k-tiles per expert
  const int NIT  = NE * KT;          // 256 pipelined iterations

  // Prologue fill (expert 0, k-tile 0).
  async_b128(aD[0], aS);
  async_b128(bD0[0], bS0);
  async_b128(bD1[0], bS1);

  for (int it = 0; it < NIT; ++it) {
    wait_async0();
    __syncthreads();
    if (it + 1 < NIT) {
      const int e2  = (it + 1) >> 6;
      const int kt2 = (it + 1) & (KT - 1);
      const size_t ao = (size_t)e2 * T_TOK * DF + (size_t)kt2 * 32;
      const size_t bo = (size_t)e2 * DF * DM + (size_t)kt2 * 32;
      const int b = (it + 1) & 1;
      async_b128(aD[b],  aS  + ao);
      async_b128(bD0[b], bS0 + bo);
      async_b128(bD1[b], bS1 + bo);
    }
    const int b = it & 1;
    Frag a[2], bb[2];
    for (int sm = 0; sm < 2; ++sm) {
      const int m = wm * 32 + sm * 16 + lr;
      a[sm].q[0] = *(const uint4*)&As[b][m * 40 + half * 8];
      a[sm].q[1] = *(const uint4*)&As[b][m * 40 + 16 + half * 8];
    }
    for (int sn = 0; sn < 2; ++sn) {
      const int n = wn * 32 + sn * 16 + lr;
      bb[sn].q[0] = *(const uint4*)&Bs[b][n * 40 + half * 8];
      bb[sn].q[1] = *(const uint4*)&Bs[b][n * 40 + 16 + half * 8];
    }
    for (int sm = 0; sm < 2; ++sm)
      for (int sn = 0; sn < 2; ++sn)
        tacc[sm][sn] = __builtin_amdgcn_wmma_f32_16x16x32_bf16(
            false, a[sm].v, false, bb[sn].v, (short)0, tacc[sm][sn], false, false);

    if ((it & (KT - 1)) == (KT - 1)) {           // expert boundary (uniform)
      const int e = it >> 6;
      for (int sm = 0; sm < 2; ++sm)
        for (int sn = 0; sn < 2; ++sn)
          for (int r = 0; r < 8; ++r) {
            const int rowL = wm * 32 + sm * 16 + r + half * 8;
            facc[sm][sn][r] += cwl[rowL * 4 + e] * tacc[sm][sn][r];
            tacc[sm][sn][r] = 0.f;
          }
    }
  }

  for (int sm = 0; sm < 2; ++sm)
    for (int sn = 0; sn < 2; ++sn)
      for (int r = 0; r < 8; ++r) {
        const int rowL = wm * 32 + sm * 16 + r + half * 8;
        const int col  = n0 + wn * 32 + sn * 16 + lr;
        out[(size_t)(t0 + rowL) * DM + col] = facc[sm][sn][r];
      }
}

// ---------------------------------------------------------------------------
extern "C" void kernel_launch(void* const* d_in, const int* in_sizes, int n_in,
                              void* d_out, int out_size, void* d_ws, size_t ws_size,
                              hipStream_t stream) {
  const float* x   = (const float*)d_in[0];
  const float* wg  = (const float*)d_in[1];
  const float* wgp = (const float*)d_in[2];
  const float* wup = (const float*)d_in[3];
  const float* wdp = (const float*)d_in[4];
  float* out = (float*)d_out;

  char* ws = (char*)d_ws;
  size_t off = 0;
  float* cw = (float*)(ws + off);                 off += (size_t)T_TOK * NE * 4;
  unsigned short* xbf  = (unsigned short*)(ws + off); off += (size_t)T_TOK * DM * 2;
  unsigned short* hbf  = (unsigned short*)(ws + off); off += (size_t)NE * T_TOK * DF * 2;
  unsigned short* wgt  = (unsigned short*)(ws + off); off += (size_t)NE * DM * DF * 2;
  unsigned short* wut  = (unsigned short*)(ws + off); off += (size_t)NE * DM * DF * 2;
  unsigned short* wdt  = (unsigned short*)(ws + off); off += (size_t)NE * DF * DM * 2;

  moe_router<<<T_TOK / 8, 256, 0, stream>>>(x, wg, cw);
  cvt_bf16<<<(T_TOK * DM / 2 + 255) / 256, 256, 0, stream>>>(
      x, (unsigned int*)xbf, T_TOK * DM / 2);
  // gate/up: src [K=DM][N=DF] -> [N][K];  down: src [K=DF][N=DM] -> [N][K]
  wt_cvt_t<<<dim3(DF / 64, DM / 64, NE), 256, 0, stream>>>(wgp, wgt, DM, DF);
  wt_cvt_t<<<dim3(DF / 64, DM / 64, NE), 256, 0, stream>>>(wup, wut, DM, DF);
  wt_cvt_t<<<dim3(DM / 64, DF / 64, NE), 256, 0, stream>>>(wdp, wdt, DF, DM);

  moe_gateup<<<dim3(DF / 128, T_TOK / 64, NE), 256, 0, stream>>>(xbf, wgt, wut, hbf);
  moe_down<<<dim3(DM / 128, T_TOK / 64), 256, 0, stream>>>(hbf, wdt, cw, out);
}
